// PrimitiveBank_15221364097077
// MI455X (gfx1250) — matrix operand order repaired and checked by reading, not compiled, not run on hardware
//
#include <hip/hip_runtime.h>

// ---------------------------------------------------------------------------
// PrimitiveBank (top-k low-rank mixture) for MI455X / gfx1250, wave32 + WMMA.
//
// Math: out = x @ Ahat @ Bhat + c, where
//   Ahat [k, p*32+r] = A[idx_p, k, r] * latent_scale[r]         (2048 x 256, f16)
//   BhatT[o, p*32+r] = top_w[p] * B[idx_p, r, o]                (2048 x 256, f16)
//   c[o]             = sum_p top_w[p] * sum_r latent_bias[r]*B[idx_p,r,o]
// GEMM1 computes U = x@Ahat; GEMM2 computes out^T = BhatT @ U^T so that both
// the U->LDS park and the output epilogue are contiguous b128 accesses.
// 34.4 GFLOP over ~272 MB HBM traffic -> memory-bound on v_wmma_f32_16x16x32_f16.
// ---------------------------------------------------------------------------

typedef _Float16 f16;
typedef __attribute__((ext_vector_type(16))) _Float16 v16h;
typedef __attribute__((ext_vector_type(8)))  _Float16 v8h;
typedef __attribute__((ext_vector_type(8)))  float    v8f;
typedef __attribute__((ext_vector_type(4)))  float    v4f;

#define D_IN   2048
#define D_OUT  2048
#define RANK   32
#define KSEL   8
#define RC     (KSEL * RANK)        // 256 = concatenated rank dim
#define M_BLK  64                   // tokens per block
#define SX     40                   // LDS halves/row for x chunk (32 + 8 pad)
#define SUT    72                   // LDS halves/row for U^T (64 tokens + 8 pad)
#define LDS_HALVES (RC * SUT)       // 18432 halves = 36864 B (x ping-pong fits inside)

// workspace layout (bytes)
#define WS_IDX   0                                  // 8 * int
#define WS_TW    32                                 // 8 * float
#define WS_C     256                                // 2048 * float
#define WS_AHAT  8448                               // 2048*256 f16 (1 MB)
#define WS_BHAT  (8448 + D_IN * RC * 2)             // 2048*256 f16 (1 MB, transposed)

static __device__ __forceinline__ v16h cat8(v8h lo, v8h hi) {
  v16h a;
  #pragma unroll
  for (int i = 0; i < 8; ++i) { a[i] = lo[i]; a[8 + i] = hi[i]; }
  return a;
}

// ---------------------------------------------------------------------- prep
__global__ void pb_topk_kernel(const float* __restrict__ w,
                               int* __restrict__ idx, float* __restrict__ tw) {
  if (threadIdx.x == 0) {
    float vals[64];
    for (int i = 0; i < 64; ++i) vals[i] = w[i];
    float sum = 0.f;
    for (int j = 0; j < KSEL; ++j) {
      int best = 0; float bv = -1e30f;
      for (int i = 0; i < 64; ++i) if (vals[i] > bv) { bv = vals[i]; best = i; }
      idx[j] = best; tw[j] = bv; vals[best] = -1e30f; sum += bv;
    }
    float inv = 1.0f / (sum + 1e-8f);
    for (int j = 0; j < KSEL; ++j) tw[j] *= inv;
  }
}

__global__ void pb_build_ahat_kernel(const float* __restrict__ A,
                                     const float* __restrict__ lscale,
                                     const int* __restrict__ idx,
                                     f16* __restrict__ Ahat) {
  int t = blockIdx.x * blockDim.x + threadIdx.x;   // 2048*8 threads
  int k = t >> 3, p = t & 7;
  const float* src = A + ((size_t)idx[p] * D_IN + k) * RANK;
  f16* dst = Ahat + (size_t)k * RC + p * RANK;
  #pragma unroll
  for (int r0 = 0; r0 < RANK; r0 += 8) {
    v8h h;
    #pragma unroll
    for (int i = 0; i < 8; ++i) h[i] = (f16)(src[r0 + i] * lscale[r0 + i]);
    *(v8h*)(dst + r0) = h;
  }
}

__global__ void pb_build_bhatT_kernel(const float* __restrict__ B,
                                      const int* __restrict__ idx,
                                      const float* __restrict__ tw,
                                      f16* __restrict__ BhatT) {
  int t = blockIdx.x * blockDim.x + threadIdx.x;   // 8*2048 threads
  int p = t >> 11;                                 // 0..7
  int o = t & (D_OUT - 1);                         // 0..2047
  const float* src = B + (size_t)idx[p] * RANK * D_OUT + o;
  float w = tw[p];
  f16* dst = BhatT + (size_t)o * RC + p * RANK;
  #pragma unroll
  for (int r0 = 0; r0 < RANK; r0 += 8) {
    v8h h;
    #pragma unroll
    for (int i = 0; i < 8; ++i) h[i] = (f16)(src[(size_t)(r0 + i) * D_OUT] * w);
    *(v8h*)(dst + r0) = h;
  }
}

__global__ void pb_build_c_kernel(const float* __restrict__ B,
                                  const int* __restrict__ idx,
                                  const float* __restrict__ tw,
                                  const float* __restrict__ lbias,
                                  float* __restrict__ cvec) {
  int o = blockIdx.x * blockDim.x + threadIdx.x;   // 2048 threads
  float s = 0.f;
  for (int p = 0; p < KSEL; ++p) {
    const float* src = B + (size_t)idx[p] * RANK * D_OUT + o;
    float acc = 0.f;
    #pragma unroll
    for (int r = 0; r < RANK; ++r) acc += lbias[r] * src[(size_t)r * D_OUT];
    s += tw[p] * acc;
  }
  cvec[o] = s;
}

// ------------------------------------------------------------------ main GEMM
__global__ __launch_bounds__(256)
void pb_fused_gemm_kernel(const float* __restrict__ x,
                          const f16* __restrict__ Ahat,
                          const f16* __restrict__ BhatT,
                          const float* __restrict__ cvec,
                          float* __restrict__ out) {
  __shared__ __attribute__((aligned(32))) f16 lds[LDS_HALVES];

  const int tid  = threadIdx.x;
  const int lane = tid & 31;
  const int wv   = tid >> 5;            // wave 0..7
  const int row0 = blockIdx.x * M_BLK;  // first token of this block

  const int mrow = lane & 15;           // A-frag row within a 16x32 tile
  const int kh   = (lane >> 4) * 8;     // A-frag K sub-offset (ISA layout)

  // ---- GEMM1: U(64x256) = x(64x2048) @ Ahat(2048x256); wave owns 32 U-cols.
  // Ping-pong x staging in LDS: one barrier per K-step, next chunk's global
  // loads issued before the barrier so their latency overlaps the WMMAs.
  v8f acc[4][2] = {};
  const int xr = tid >> 2;              // 0..63 row in tile
  const int xc = (tid & 3) * 8;         // 0,8,16,24 col in 32-wide chunk
  const float* xsrc = x + (size_t)(row0 + xr) * D_IN + xc;

  v4f f0 = __builtin_nontemporal_load((const v4f*)(xsrc));
  v4f f1 = __builtin_nontemporal_load((const v4f*)(xsrc + 4));

  for (int ks = 0; ks < D_IN / 32; ++ks) {
    f16* buf = &lds[(ks & 1) * (M_BLK * SX)];
    v8h h;
    #pragma unroll
    for (int i = 0; i < 4; ++i) { h[i] = (f16)f0[i]; h[4 + i] = (f16)f1[i]; }
    *(v8h*)(&buf[xr * SX + xc]) = h;

    if (ks < D_IN / 32 - 1) {           // issue next chunk before the barrier
      const float* nsrc = xsrc + (ks + 1) * 32;
      f0 = __builtin_nontemporal_load((const v4f*)(nsrc));
      f1 = __builtin_nontemporal_load((const v4f*)(nsrc + 4));
      __builtin_prefetch(nsrc + 32, 0, 1);   // global_prefetch_b8
    }
    __syncthreads();

    v16h afrag[4];
    #pragma unroll
    for (int mt = 0; mt < 4; ++mt) {
      const f16* p = &buf[(mt * 16 + mrow) * SX + kh];
      afrag[mt] = cat8(*(const v8h*)(p), *(const v8h*)(p + 16));
    }
    #pragma unroll
    for (int nt = 0; nt < 2; ++nt) {
      v16h bfrag;
      #pragma unroll
      for (int s = 0; s < 16; ++s)
        bfrag[s] = Ahat[(size_t)(ks * 32 + kh + (s & 7) + ((s >> 3) << 4)) * RC + wv * 32 + nt * 16 + mrow];
      #pragma unroll
      for (int mt = 0; mt < 4; ++mt)
        acc[mt][nt] = __builtin_amdgcn_wmma_f32_16x16x32_f16(
            false, afrag[mt], false, bfrag, (short)0, acc[mt][nt], false, false);
    }
  }
  __syncthreads();                      // last reads of x buffers done

  // ---- park U^T (f16) in LDS: D-frag's 8 VGPRs = 8 consecutive tokens of one
  // U^T row -> single ds_store_b128 per tile.
  #pragma unroll
  for (int mt = 0; mt < 4; ++mt)
    #pragma unroll
    for (int nt = 0; nt < 2; ++nt) {
      const int n  = wv * 32 + nt * 16 + (lane & 15);   // rank-cat index (U^T row)
      const int m0 = mt * 16 + ((lane >> 4) << 3);      // token (U^T col base)
      v8h h;
      #pragma unroll
      for (int v = 0; v < 8; ++v) h[v] = (f16)acc[mt][nt][v];
      *(v8h*)(&lds[n * SUT + m0]) = h;
    }
  __syncthreads();

  // ---- GEMM2 (transposed): out^T(2048x64) = BhatT(2048x256) @ U^T(256x64).
  // Wave owns out-col tiles [wv*256, wv*256+256); tokens give 4 N-tiles.
  for (int mt = 0; mt < 16; ++mt) {
    const int mcol0 = wv * 256 + mt * 16;
    v8f oacc[4] = {};
    #pragma unroll
    for (int ks = 0; ks < RC / 32; ++ks) {
      const f16* ap = BhatT + (size_t)(mcol0 + mrow) * RC + ks * 32 + kh;
      v16h afrag = cat8(*(const v8h*)(ap), *(const v8h*)(ap + 16));
      #pragma unroll
      for (int nt = 0; nt < 4; ++nt) {
        v16h bfrag;
        #pragma unroll
        for (int s = 0; s < 16; ++s)
          bfrag[s] = lds[(ks * 32 + kh + (s & 7) + ((s >> 3) << 4)) * SUT + nt * 16 + mrow];
        oacc[nt] = __builtin_amdgcn_wmma_f32_16x16x32_f16(
            false, afrag, false, bfrag, (short)0, oacc[nt], false, false);
      }
    }
    // epilogue: D-frag's 8 VGPRs = 8 consecutive out columns of one token.
    const int colb = mcol0 + ((lane >> 4) << 3);
    v4f cv0 = *(const v4f*)(cvec + colb);
    v4f cv1 = *(const v4f*)(cvec + colb + 4);
    #pragma unroll
    for (int nt = 0; nt < 4; ++nt) {
      const int token = row0 + nt * 16 + (lane & 15);
      float* op = out + (size_t)token * D_OUT + colb;
      v4f r0, r1;
      #pragma unroll
      for (int i = 0; i < 4; ++i) { r0[i] = oacc[nt][i] + cv0[i]; r1[i] = oacc[nt][4 + i] + cv1[i]; }
      __builtin_nontemporal_store(r0, (v4f*)op);
      __builtin_nontemporal_store(r1, (v4f*)(op + 4));
    }
  }
}

// ------------------------------------------------------------------- launcher
extern "C" void kernel_launch(void* const* d_in, const int* in_sizes, int n_in,
                              void* d_out, int out_size, void* d_ws, size_t ws_size,
                              hipStream_t stream) {
  const float* x      = (const float*)d_in[0];
  const float* wts    = (const float*)d_in[1];
  const float* A      = (const float*)d_in[2];
  const float* B      = (const float*)d_in[3];
  const float* lscale = (const float*)d_in[4];
  const float* lbias  = (const float*)d_in[5];
  float* out = (float*)d_out;

  char* ws = (char*)d_ws;
  int*   idx   = (int*)(ws + WS_IDX);
  float* tw    = (float*)(ws + WS_TW);
  float* cvec  = (float*)(ws + WS_C);
  f16*   Ahat  = (f16*)(ws + WS_AHAT);
  f16*   BhatT = (f16*)(ws + WS_BHAT);

  const int T = in_sizes[0] / D_IN;   // 16384 tokens

  pb_topk_kernel<<<1, 32, 0, stream>>>(wts, idx, tw);
  pb_build_ahat_kernel<<<(D_IN * KSEL) / 256, 256, 0, stream>>>(A, lscale, idx, Ahat);
  pb_build_bhatT_kernel<<<(KSEL * D_OUT) / 256, 256, 0, stream>>>(B, idx, tw, BhatT);
  pb_build_c_kernel<<<D_OUT / 256, 256, 0, stream>>>(B, idx, tw, lbias, cvec);
  pb_fused_gemm_kernel<<<T / M_BLK, 256, 0, stream>>>(x, Ahat, BhatT, cvec, out);
}
